// EntropyByteLatentTransformer_13649406067283
// MI455X (gfx1250) — matrix operand order repaired and checked by reading, not compiled, not run on hardware
//
#include <hip/hip_runtime.h>
#include <math.h>

// ---------------------------------------------------------------------------
// Problem constants (from the reference)
// ---------------------------------------------------------------------------
#define B_    4
#define L_    2048
#define D_    512
#define H_    8
#define HD_   64
#define NL_   4
#define V_    258
#define W_    8
#define M_    (B_ * L_)      // 8192 rows of activations
#define D3_   (3 * D_)       // 1536
#define D4_   (4 * D_)       // 2048

#define LSTR  36             // LDS row stride in ushorts (72B: b64-store aligned,
                             // conflict-free 16-lane fragment gathers)

// ---------------------------------------------------------------------------
// CDNA5 WMMA types (wave32). bf16 16x16x32, f32 accumulate.
// ---------------------------------------------------------------------------
typedef __attribute__((ext_vector_type(16))) __bf16 v16bf;
typedef __attribute__((ext_vector_type(8)))  float  v8f;

union FragU { unsigned int u[8]; v16bf v; };

__device__ __forceinline__ unsigned short f2bf(float f) {
  unsigned int u = __builtin_bit_cast(unsigned int, f);
  unsigned int r = (u + 0x7FFFu + ((u >> 16) & 1u)) >> 16;  // round-to-nearest-even
  return (unsigned short)r;
}

__device__ __forceinline__ unsigned int pack2bf(float a, float b) {
  return (unsigned)f2bf(a) | ((unsigned)f2bf(b) << 16);
}

// ---------------------------------------------------------------------------
// Kernel 1: token embedding + sliding-window entropy feature
// ---------------------------------------------------------------------------
__global__ __launch_bounds__(256)
void embed_entropy_kernel(const float* __restrict__ emb,
                          const float* __restrict__ ent_w,
                          const float* __restrict__ ent_b,
                          const int*   __restrict__ x,
                          float*       __restrict__ h)
{
  const int i = blockIdx.x;           // flat (b, l)
  const int b = i / L_;
  const int l = i - b * L_;
  __shared__ float ent_s;

  if (threadIdx.x == 0) {
    float ent = 0.f;
    if (l <= L_ - W_) {               // positions [0, L-W] get entropy, rest 0
      int w[W_];
      #pragma unroll
      for (int j = 0; j < W_; ++j) w[j] = x[b * L_ + l + j];
      float e = 0.f;
      #pragma unroll
      for (int j = 0; j < W_; ++j) {
        if (w[j] >= 256) continue;    // one_hot(256 classes): 256/257 contribute 0
        int c = 0; bool first = true;
        #pragma unroll
        for (int k = 0; k < W_; ++k) {
          if (w[k] == w[j]) { ++c; if (k < j) first = false; }
        }
        if (first) {
          float p = (float)c * (1.0f / (float)W_);
          e -= p * log2f(p + 1e-10f);
        }
      }
      ent = e;
    }
    ent_s = ent;
  }
  __syncthreads();

  const int tok = x[i];
  const float ent = ent_s;
  for (int d = threadIdx.x; d < D_; d += 256)
    h[(size_t)i * D_ + d] = emb[(size_t)tok * D_ + d] + ent * ent_w[d] + ent_b[d];
}

// ---------------------------------------------------------------------------
// Kernel 2: LayerNorm over D=512, one block per row
// ---------------------------------------------------------------------------
__global__ __launch_bounds__(128)
void ln_kernel(const float* __restrict__ x,
               const float* __restrict__ s,
               const float* __restrict__ bgain,
               float*       __restrict__ y)
{
  const int row = blockIdx.x;
  const int tid = threadIdx.x;
  const float* xr = x + (size_t)row * D_;
  __shared__ float sh[128];

  float v[4];
  float sum = 0.f;
  #pragma unroll
  for (int i = 0; i < 4; ++i) { v[i] = xr[tid + 128 * i]; sum += v[i]; }
  sh[tid] = sum; __syncthreads();
  #pragma unroll
  for (int off = 64; off > 0; off >>= 1) {
    if (tid < off) sh[tid] += sh[tid + off];
    __syncthreads();
  }
  const float mu = sh[0] * (1.0f / D_);
  __syncthreads();

  float ss = 0.f;
  #pragma unroll
  for (int i = 0; i < 4; ++i) { float d = v[i] - mu; ss += d * d; }
  sh[tid] = ss; __syncthreads();
  #pragma unroll
  for (int off = 64; off > 0; off >>= 1) {
    if (tid < off) sh[tid] += sh[tid + off];
    __syncthreads();
  }
  const float rstd = rsqrtf(sh[0] * (1.0f / D_) + 1e-5f);

  #pragma unroll
  for (int i = 0; i < 4; ++i) {
    int d = tid + 128 * i;
    y[(size_t)row * D_ + d] = (v[i] - mu) * rstd * s[d] + bgain[d];
  }
}

// ---------------------------------------------------------------------------
// Kernel 3: WMMA bf16 GEMM   C[M,N] = act(A[M,K] @ Wt[N,K]^T + bias) (+ R)
// 256 threads = 8 waves; block tile 128x128; each wave owns 32x64 (2x4 WMMA
// tiles -> 8 wmma per K-step). K-step 32, double-buffered LDS, float4 global
// loads packed to bf16 pairs and stored with ds_store_b64.
// act: 0 = none, 1 = exact GELU.
// ---------------------------------------------------------------------------
__global__ __launch_bounds__(256)
void gemm_wmma(const float* __restrict__ A,
               const float* __restrict__ Wt,
               const float* __restrict__ bias,
               const float* __restrict__ R,
               float*       __restrict__ C,
               int M, int N, int K, int act)
{
  __shared__ unsigned short As[2][128 * LSTR];  // [buf][row][k]
  __shared__ unsigned short Bs[2][128 * LSTR];  // [buf][n][k]

  const int tid  = threadIdx.x;
  const int lane = tid & 31;
  const int wid  = tid >> 5;               // 0..7
  const int widm = wid & 3;                // m sub-tile (32 rows each)
  const int widn = wid >> 2;               // n sub-tile (64 cols each)
  const int n16  = lane & 15;
  const int hi   = lane >> 4;              // EXEC half select per ISA layouts
  const int m0   = blockIdx.y * 128;
  const int n0   = blockIdx.x * 128;
  const int nk   = K >> 5;                 // number of 32-wide K tiles

  v8f acc[2][4] = {};                      // [a: 16-row subtile][s: 16-col subtile]

  // Stage one 128x32 tile of A and W into LDS buffer `buf` (f32 -> bf16).
  auto stage = [&](int buf, int k0) {
    #pragma unroll
    for (int i = 0; i < 4; ++i) {
      int e4 = tid + i * 256;              // 1024 float4 groups
      int r  = e4 >> 3;                    // 0..127
      int c  = (e4 & 7) * 4;               // 0,4,...,28
      float4 fa = *(const float4*)&A[(size_t)(m0 + r) * K + k0 + c];
      uint2  pa; pa.x = pack2bf(fa.x, fa.y); pa.y = pack2bf(fa.z, fa.w);
      *(uint2*)&As[buf][r * LSTR + c] = pa;
      int wr = n0 + r;
      float4 fb = make_float4(0.f, 0.f, 0.f, 0.f);
      if (wr < N) fb = *(const float4*)&Wt[(size_t)wr * K + k0 + c];
      uint2  pb; pb.x = pack2bf(fb.x, fb.y); pb.y = pack2bf(fb.z, fb.w);
      *(uint2*)&Bs[buf][r * LSTR + c] = pb;
    }
    if (k0 + 32 < K) {                     // gfx1250 global_prefetch_b8 of tile after next
      int pr = tid >> 1;
      __builtin_prefetch(&A[(size_t)(m0 + pr) * K + k0 + 32], 0, 1);
      int wr2 = n0 + pr;
      if (wr2 < N) __builtin_prefetch(&Wt[(size_t)wr2 * K + k0 + 32], 0, 1);
    }
  };

  stage(0, 0);

  for (int kt = 0; kt < nk; ++kt) {
    __syncthreads();                       // stores of buf p visible; WAR on buf p^1 handled
    if (kt + 1 < nk) stage((kt + 1) & 1, (kt + 1) * 32);
    const int p = kt & 1;

    // Two A fragments (rows widm*32 + {0,16}); lane.m = lane&15, hi picks K group.
    FragU af[2];
    #pragma unroll
    for (int a = 0; a < 2; ++a)
      #pragma unroll
      for (int v = 0; v < 8; ++v) {
        int kb = ((v & 4) << 2) + (v & 3) * 2 + hi * 8;   // {0|16} + 2*(v&3) + 8*hi
        af[a].u[v] = *(const unsigned int*)&As[p][(widm * 32 + a * 16 + n16) * LSTR + kb];
      }
    #pragma unroll
    for (int s = 0; s < 4; ++s) {
      // B fragment: lane.n = lane&15; VGPR v holds K pair {0|16} + 2*(v&3) + 8*hi (same as A).
      FragU bf_;
      #pragma unroll
      for (int v = 0; v < 8; ++v)
        bf_.u[v] = *(const unsigned int*)&Bs[p][(widn * 64 + s * 16 + n16) * LSTR + (((v & 4) << 2) + (v & 3) * 2 + hi * 8)];
      #pragma unroll
      for (int a = 0; a < 2; ++a)
        acc[a][s] = __builtin_amdgcn_wmma_f32_16x16x32_bf16(
            false, af[a].v, false, bf_.v, (short)0, acc[a][s], false, false);
    }
  }

  // Epilogue: C layout — lane (n,hi), VGPR r -> C[r + 8*hi][n].
  #pragma unroll
  for (int s = 0; s < 4; ++s) {
    int col = n0 + widn * 64 + s * 16 + n16;
    if (col >= N) continue;
    float bv = bias ? bias[col] : 0.f;
    #pragma unroll
    for (int a = 0; a < 2; ++a)
      #pragma unroll
      for (int r = 0; r < 8; ++r) {
        int row = m0 + widm * 32 + a * 16 + r + hi * 8;
        float val = acc[a][s][r] + bv;
        if (act == 1) val = 0.5f * val * (1.f + erff(val * 0.70710678118654752f));
        if (R) val += R[(size_t)row * N + col];
        C[(size_t)row * N + col] = val;
      }
  }
}

// ---------------------------------------------------------------------------
// Kernel 4: flash attention with WMMA. One wave per (b, h, 16-query tile).
// qkv: [B, L, 3D] with q|k|v contiguous, head h at column h*64 of each block.
// ---------------------------------------------------------------------------
__global__ __launch_bounds__(32)
void attn_wmma(const float* __restrict__ qkv, float* __restrict__ O)
{
  __shared__ unsigned short Ps[16 * 34];    // probs, [query m][key kk]
  __shared__ unsigned short VsT[64 * 34];   // V transposed: [hd][key]

  const int lane = threadIdx.x;
  const int n16  = lane & 15;
  const int hi   = lane >> 4;
  const int q0   = blockIdx.x * 16;
  const int h    = blockIdx.y;
  const int b    = blockIdx.z;
  const size_t base = (size_t)b * L_ * D3_;

  // Q fragments for K-dim 64 = two 16x32 chunks; fold in 1/sqrt(HD)=0.125.
  const float* qp = qkv + base + (size_t)(q0 + n16) * D3_ + h * HD_;
  FragU qa[2];
  #pragma unroll
  for (int c = 0; c < 2; ++c)
    #pragma unroll
    for (int v = 0; v < 8; ++v) {
      int kk = c * 32 + ((v & 4) << 2) + (v & 3) * 2 + hi * 8;
      float2 f = *(const float2*)&qp[kk];                 // 8B aligned (kk even)
      qa[c].u[v] = pack2bf(f.x * 0.125f, f.y * 0.125f);
    }

  float mrow[8], lrow[8];
  #pragma unroll
  for (int r = 0; r < 8; ++r) { mrow[r] = -1e30f; lrow[r] = 0.f; }
  v8f oacc[4] = {};

  for (int jt = 0; jt < L_ / 32; ++jt) {
    const int kb = jt * 32;
    __syncthreads();                                      // WAR on VsT/Ps

    // Stage V tile (32 keys x 64 hd) transposed into LDS; lane <-> key.
    {
      const float* vp = qkv + base + (size_t)(kb + lane) * D3_ + 2 * D_ + h * HD_;
      #pragma unroll
      for (int d4 = 0; d4 < HD_; d4 += 4) {
        float4 f = *(const float4*)&vp[d4];               // 16B aligned
        VsT[(d4 + 0) * 34 + lane] = f2bf(f.x);
        VsT[(d4 + 1) * 34 + lane] = f2bf(f.y);
        VsT[(d4 + 2) * 34 + lane] = f2bf(f.z);
        VsT[(d4 + 3) * 34 + lane] = f2bf(f.w);
      }
    }

    // Scores: two 16x16 tiles (32 keys), each = 2 chained WMMAs over K=64.
    v8f sc[2];
    #pragma unroll
    for (int t = 0; t < 2; ++t) {
      const float* kp = qkv + base + (size_t)(kb + t * 16 + n16) * D3_ + D_ + h * HD_;
      FragU kf[2];
      #pragma unroll
      for (int c = 0; c < 2; ++c)
        #pragma unroll
        for (int v = 0; v < 8; ++v) {
          int kk = c * 32 + ((v & 4) << 2) + (v & 3) * 2 + hi * 8;              // B layout: K pairs per VGPR
          float2 f = *(const float2*)&kp[kk];
          kf[c].u[v] = pack2bf(f.x, f.y);
        }
      v8f z = {};
      z = __builtin_amdgcn_wmma_f32_16x16x32_bf16(false, qa[1].v, false, kf[1].v, (short)0, z, false, false);
      z = __builtin_amdgcn_wmma_f32_16x16x32_bf16(false, qa[0].v, false, kf[0].v, (short)0, z, false, false);
      sc[t] = z;
    }

    // Online softmax. Score row m = r + 8*hi lives across the 16 lanes of one
    // EXEC half -> xor masks 1,2,4,8 reduce exactly over keys of that row.
    #pragma unroll
    for (int r = 0; r < 8; ++r) {
      float t = fmaxf(sc[0][r], sc[1][r]);
      t = fmaxf(t, __shfl_xor(t, 1));
      t = fmaxf(t, __shfl_xor(t, 2));
      t = fmaxf(t, __shfl_xor(t, 4));
      t = fmaxf(t, __shfl_xor(t, 8));
      float mnew  = fmaxf(mrow[r], t);
      float scale = expf(mrow[r] - mnew);
      float p0 = expf(sc[0][r] - mnew);
      float p1 = expf(sc[1][r] - mnew);
      float rs = p0 + p1;
      rs += __shfl_xor(rs, 1);
      rs += __shfl_xor(rs, 2);
      rs += __shfl_xor(rs, 4);
      rs += __shfl_xor(rs, 8);
      lrow[r] = lrow[r] * scale + rs;
      mrow[r] = mnew;
      #pragma unroll
      for (int s = 0; s < 4; ++s) oacc[s][r] *= scale;
      // C-layout -> LDS so we can regather P in A-layout.
      Ps[(r + hi * 8) * 34 + n16]      = f2bf(p0);
      Ps[(r + hi * 8) * 34 + 16 + n16] = f2bf(p1);
    }
    __syncthreads();

    // P as 16x32 A fragment, V as 32x16 B fragments; O += P @ V.
    FragU pf;
    #pragma unroll
    for (int v = 0; v < 8; ++v) {
      int kk = ((v & 4) << 2) + (v & 3) * 2 + hi * 8;
      pf.u[v] = *(const unsigned int*)&Ps[n16 * 34 + kk];
    }
    #pragma unroll
    for (int s = 0; s < 4; ++s) {
      FragU vf;
      #pragma unroll
      for (int v = 0; v < 8; ++v)
        vf.u[v] = *(const unsigned int*)&VsT[(s * 16 + n16) * 34 + (((v & 4) << 2) + (v & 3) * 2 + hi * 8)];
      oacc[s] = __builtin_amdgcn_wmma_f32_16x16x32_bf16(
          false, pf.v, false, vf.v, (short)0, oacc[s], false, false);
    }
  }

  // Normalize and write O[b, q, h*64 + hd].
  #pragma unroll
  for (int s = 0; s < 4; ++s)
    #pragma unroll
    for (int r = 0; r < 8; ++r) {
      int row = q0 + r + hi * 8;
      O[((size_t)b * L_ + row) * D_ + h * HD_ + s * 16 + n16] = oacc[s][r] / lrow[r];
    }
}

// ---------------------------------------------------------------------------
// Driver
// ---------------------------------------------------------------------------
extern "C" void kernel_launch(void* const* d_in, const int* in_sizes, int n_in,
                              void* d_out, int out_size, void* d_ws, size_t ws_size,
                              hipStream_t stream) {
  (void)in_sizes; (void)n_in; (void)out_size; (void)ws_size;
  const float* emb   = (const float*)d_in[0];
  const float* ent_w = (const float*)d_in[1];
  const float* ent_b = (const float*)d_in[2];
  const float* qkv_w = (const float*)d_in[3];
  const float* qkv_b = (const float*)d_in[4];
  const float* ao_w  = (const float*)d_in[5];
  const float* ao_b  = (const float*)d_in[6];
  const float* ln1_s = (const float*)d_in[7];
  const float* ln1_b = (const float*)d_in[8];
  const float* ln2_s = (const float*)d_in[9];
  const float* ln2_b = (const float*)d_in[10];
  const float* ff1_w = (const float*)d_in[11];
  const float* ff1_b = (const float*)d_in[12];
  const float* ff2_w = (const float*)d_in[13];
  const float* ff2_b = (const float*)d_in[14];
  const float* out_w = (const float*)d_in[15];
  const int*   x     = (const int*)d_in[16];
  // d_in[17] (patch_lengths) is dead in the reference.

  // Workspace layout (96 MB): h | g(/o) | big(qkv 48MB / ffn-hidden 64MB)
  char*  ws  = (char*)d_ws;
  float* h   = (float*)(ws);
  float* g   = (float*)(ws + (size_t)16 * 1024 * 1024);
  float* big = (float*)(ws + (size_t)32 * 1024 * 1024);
  float* o   = g;   // g is dead once qkv is computed; reuse for attention out

  embed_entropy_kernel<<<M_, 256, 0, stream>>>(emb, ent_w, ent_b, x, h);

  for (int l = 0; l < NL_; ++l) {
    ln_kernel<<<M_, 128, 0, stream>>>(h, ln1_s + l * D_, ln1_b + l * D_, g);
    gemm_wmma<<<dim3(D3_ / 128, M_ / 128), 256, 0, stream>>>(
        g, qkv_w + (size_t)l * D3_ * D_, qkv_b + (size_t)l * D3_, nullptr,
        big, M_, D3_, D_, 0);
    attn_wmma<<<dim3(L_ / 16, H_, B_), 32, 0, stream>>>(big, o);
    gemm_wmma<<<dim3(D_ / 128, M_ / 128), 256, 0, stream>>>(
        o, ao_w + (size_t)l * D_ * D_, ao_b + (size_t)l * D_, h,
        h, M_, D_, D_, 0);
    ln_kernel<<<M_, 128, 0, stream>>>(h, ln2_s + l * D_, ln2_b + l * D_, g);
    gemm_wmma<<<dim3(D4_ / 128, M_ / 128), 256, 0, stream>>>(
        g, ff1_w + (size_t)l * D4_ * D_, ff1_b + (size_t)l * D4_, nullptr,
        big, M_, D4_, D_, 1);
    gemm_wmma<<<dim3(D_ / 128, M_ / 128), 256, 0, stream>>>(
        big, ff2_w + (size_t)l * D_ * D4_, ff2_b + (size_t)l * D_, h,
        h, M_, D_, D4_, 0);
  }

  gemm_wmma<<<dim3((V_ + 127) / 128, M_ / 128), 256, 0, stream>>>(
      h, out_w, nullptr, nullptr, (float*)d_out, M_, V_, D_, 0);
}
